// GRU_Full_51900384804955
// MI455X (gfx1250) — matrix operand order repaired and checked by reading, not compiled, not run on hardware
//
#include <hip/hip_runtime.h>

// ---------------- problem constants ----------------
#define DIM   1024
#define BATCH 128
#define VOUT  8000
#define S_SRC 256
#define S_TGT 128
#define SOS_TOK 1

#define NGRP_GRU (DIM / 16)     // 64 column groups for the GRU GEMMs
#define NGRP_OUT (VOUT / 16)    // 500 column groups for the logits GEMM
#define FRAGS 6                 // ih{r,z,n} + hh{r,z,n}
#define KCH_GRU 128             // k-rows per TDM chunk (GRU):   128*6*16*2B = 24 KB
#define KCH_OUT 256             // k-rows per TDM chunk (logits): 256*16*2B  =  8 KB

// ---------------- vector types ----------------
typedef __bf16 v8bf  __attribute__((ext_vector_type(8)));
typedef __bf16 v16bf __attribute__((ext_vector_type(16)));
typedef float  v8f   __attribute__((ext_vector_type(8)));
typedef float  v4f   __attribute__((ext_vector_type(4)));
typedef unsigned int v4u __attribute__((ext_vector_type(4)));
typedef int    v4i   __attribute__((ext_vector_type(4)));
typedef int    v8i   __attribute__((ext_vector_type(8)));

union AFrag { v16bf v; v8bf h[2]; };

#if __has_builtin(__builtin_amdgcn_tensor_load_to_lds)
#define HAVE_TDM 1
#else
#define HAVE_TDM 0
#endif

__device__ __forceinline__ __bf16 f2bf(float f) {
    unsigned u = __float_as_uint(f);
    unsigned r = (u + 0x7FFFu + ((u >> 16) & 1u)) >> 16;
    unsigned short s = (unsigned short)r;
    __bf16 b;
    __builtin_memcpy(&b, &s, 2);
    return b;
}

__device__ __forceinline__ float sigmoidf_(float x) {
    return 1.0f / (1.0f + __expf(-x));
}

#if HAVE_TDM
// 1-D contiguous TDM copy: nelem bf16 elements global -> LDS.
// D# per CDNA5 ISA: group0 = {count=1, lds_addr,
// global_addr[56:0], type=2}; group1 = {data_size=2B, tensor_dim0 = tile_dim0
// = nelem, tensor_dim0_stride = nelem}; groups 2/3 unused (zero).
__device__ __forceinline__ void tdm_load_1d(unsigned ldsOff, const void* gptr, unsigned nelem) {
    unsigned long long ga = (unsigned long long)(uintptr_t)gptr;
    v4u g0;
    g0.x = 1u;                                                   // count=1, user mode
    g0.y = ldsOff;                                               // lds_addr
    g0.z = (unsigned)(ga & 0xFFFFFFFFull);                       // global_addr[31:0]
    g0.w = (unsigned)((ga >> 32) & 0x1FFFFFFull) | (2u << 30);   // [56:32] | type=2
    v8i g1;
    g1[0] = (int)(1u << 16);                    // data_size code 1 (2 bytes)
    g1[1] = (int)((nelem & 0xFFFFu) << 16);     // tensor_dim0[15:0]
    g1[2] = (int)((nelem >> 16) & 0xFFFFu);     // tensor_dim0[31:16]
    g1[3] = (int)((nelem & 0xFFFFu) << 16);     // tile_dim0 (= nelem, fits 16b)
    g1[4] = 0;                                  // tile_dim1/2 unused
    g1[5] = (int)nelem;                         // tensor_dim0_stride
    g1[6] = 0;
    g1[7] = 0;
    v4i z4 = {0, 0, 0, 0};
#if __clang_major__ >= 23
    v8i z8 = {0, 0, 0, 0, 0, 0, 0, 0};
    __builtin_amdgcn_tensor_load_to_lds(g0, g1, z4, z4, z8, 0);
#else
    __builtin_amdgcn_tensor_load_to_lds(g0, g1, z4, z4, 0);
#endif
}
#endif

// ---------------- weight prep ----------------
// GRU weights packed per 16-wide column group, k-major, 6 fragments interleaved:
// pack[((g*DIM+r)*FRAGS+f)*16+s] = W[gate][g*16+r%16][k(r,s)] ; f<3 -> W_ih gate f, f>=3 -> W_hh gate f-3.
// Each block's weight stream becomes one contiguous 192 KB region. k(r,s) = (r&~31)+8*((r>>4)&1)+16*(s>>3)+(s&7).
__global__ void pack_gru_kernel(const float* __restrict__ Wih,
                                const float* __restrict__ Whh,
                                __bf16* __restrict__ out) {
    long long i = (long long)blockIdx.x * blockDim.x + threadIdx.x;
    const long long total = (long long)NGRP_GRU * DIM * FRAGS * 16;
    if (i >= total) return;
    int c = (int)(i & 15);
    long long r = i >> 4;
    int f = (int)(r % FRAGS); r /= FRAGS;
    int kk = (int)(r & (DIM - 1));
    int k = (kk & ~31) + ((kk >> 4) & 1) * 8 + ((c & 8) << 1) + (c & 7);
    int g = (int)(r >> 10);
    const float* W = (f < 3) ? Wih : Whh;
    int gate = (f < 3) ? f : f - 3;
    int n = gate * DIM + g * 16 + (kk & 15);
    out[i] = f2bf(W[(size_t)n * DIM + k]);
}

// out_W packed: pack[(g*DIM + r)*16 + s] = out_W[(g*16+r%16)*DIM + k(r,s)], k(r,s) as for the GRU pack
__global__ void pack_out_kernel(const float* __restrict__ W, __bf16* __restrict__ out) {
    long long i = (long long)blockIdx.x * blockDim.x + threadIdx.x;
    const long long total = (long long)NGRP_OUT * DIM * 16;
    if (i >= total) return;
    int c = (int)(i & 15);
    long long r = i >> 4;
    int kk = (int)(r & (DIM - 1));
    int k = (kk & ~31) + ((kk >> 4) & 1) * 8 + ((c & 8) << 1) + (c & 7);
    int g = (int)(r >> 10);
    out[i] = f2bf(W[(size_t)(g * 16 + (kk & 15)) * DIM + k]);
}

// ---------------- init: h = 0, tok = SOS ----------------
__global__ void init_kernel(float* __restrict__ h, __bf16* __restrict__ hb,
                            int* __restrict__ tok) {
    int idx = blockIdx.x * blockDim.x + threadIdx.x;
    if (idx < BATCH * DIM) { h[idx] = 0.0f; hb[idx] = f2bf(0.0f); }
    if (idx < BATCH) tok[idx] = SOS_TOK;
}

// Load A fragments: x gathered from fp32 embedding row (cvt to bf16 inline),
// h from the bf16 recurrent state. Per-lane layout per ISA §7.12.2:
// chunks [kb..kb+7] and [kb+16..kb+23], kb = (lane>=16)*8.
__device__ __forceinline__ void load_a_frags(const float* __restrict__ xrow,
                                             const __bf16* __restrict__ hrow,
                                             int koff, AFrag& ax, AFrag& ah) {
    const float* p0 = xrow + koff;
    v4f x0 = *reinterpret_cast<const v4f*>(p0);
    v4f x1 = *reinterpret_cast<const v4f*>(p0 + 4);
    v4f x2 = *reinterpret_cast<const v4f*>(p0 + 16);
    v4f x3 = *reinterpret_cast<const v4f*>(p0 + 20);
#pragma unroll
    for (int e = 0; e < 4; ++e) {
        ax.v[e]      = f2bf(x0[e]);
        ax.v[4 + e]  = f2bf(x1[e]);
        ax.v[8 + e]  = f2bf(x2[e]);
        ax.v[12 + e] = f2bf(x3[e]);
    }
    ah.h[0] = *reinterpret_cast<const v8bf*>(hrow + koff);
    ah.h[1] = *reinterpret_cast<const v8bf*>(hrow + koff + 16);
}

// ---------------- fused GRU step (embedding gather + 6-acc WMMA + gates) ----
// grid = 64 blocks (one 16-wide h column group), block = 256 thr = 8 waves,
// wave w owns M-tile [16w,16w+16). B weights streamed via TDM double-buffer.
__global__ __launch_bounds__(256)
void gru_step_kernel(const float*  __restrict__ emb,
                     const int*    __restrict__ tokens,
                     const __bf16* __restrict__ hbf,
                     const float*  __restrict__ hf,
                     const __bf16* __restrict__ Wpack,  // [NGRP][DIM][6][16]
                     const float*  __restrict__ bIh,
                     const float*  __restrict__ bHh,
                     float*  __restrict__ hOut,
                     __bf16* __restrict__ hOutBf) {
    const int lane = threadIdx.x & 31;
    const int wave = threadIdx.x >> 5;
    const int m0 = wave * 16;
    const int n0 = blockIdx.x * 16;

    v8f aIR = {}, aIZ = {}, aIN = {}, aHR = {}, aHZ = {}, aHN = {};

    const int mrow = m0 + (lane & 15);
    const int kb   = (lane >> 4) * 8;
    const float*  xrow = emb + (size_t)tokens[mrow] * DIM;
    const __bf16* hrow = hbf + (size_t)mrow * DIM;
    const __bf16* Wg   = Wpack + (size_t)blockIdx.x * DIM * (FRAGS * 16);

#if HAVE_TDM
    __shared__ __align__(128) __bf16 smem[2][KCH_GRU * FRAGS * 16];
    const unsigned ldsA = (unsigned)(uintptr_t)&smem[0][0];
    const unsigned ldsB = (unsigned)(uintptr_t)&smem[1][0];
    const unsigned CHE = KCH_GRU * FRAGS * 16;   // elements per chunk (12288)

    if (wave == 0) {
        tdm_load_1d(ldsA, Wg, CHE);
        __builtin_amdgcn_s_wait_tensorcnt(0);
    }
    __syncthreads();

    for (int c = 0; c < DIM / KCH_GRU; ++c) {
        const __bf16* curB = &smem[c & 1][0];
        if (wave == 0 && c + 1 < DIM / KCH_GRU)
            tdm_load_1d((c & 1) ? ldsA : ldsB, Wg + (size_t)(c + 1) * CHE, CHE);

#pragma unroll
        for (int kr = 0; kr < KCH_GRU; kr += 32) {
            const int k0 = c * KCH_GRU + kr;
            AFrag ax, ah;
            load_a_frags(xrow, hrow, k0 + kb, ax, ah);
            const __bf16* lrow = curB + (size_t)(kr + lane) * (FRAGS * 16);
            v16bf bIRf = *reinterpret_cast<const v16bf*>(lrow);
            v16bf bIZf = *reinterpret_cast<const v16bf*>(lrow + 16);
            v16bf bINf = *reinterpret_cast<const v16bf*>(lrow + 32);
            v16bf bHRf = *reinterpret_cast<const v16bf*>(lrow + 48);
            v16bf bHZf = *reinterpret_cast<const v16bf*>(lrow + 64);
            v16bf bHNf = *reinterpret_cast<const v16bf*>(lrow + 80);
            aIR = __builtin_amdgcn_wmma_f32_16x16x32_bf16(false, ax.v, false, bIRf, (short)0, aIR, false, false);
            aIZ = __builtin_amdgcn_wmma_f32_16x16x32_bf16(false, ax.v, false, bIZf, (short)0, aIZ, false, false);
            aIN = __builtin_amdgcn_wmma_f32_16x16x32_bf16(false, ax.v, false, bINf, (short)0, aIN, false, false);
            aHR = __builtin_amdgcn_wmma_f32_16x16x32_bf16(false, ah.v, false, bHRf, (short)0, aHR, false, false);
            aHZ = __builtin_amdgcn_wmma_f32_16x16x32_bf16(false, ah.v, false, bHZf, (short)0, aHZ, false, false);
            aHN = __builtin_amdgcn_wmma_f32_16x16x32_bf16(false, ah.v, false, bHNf, (short)0, aHN, false, false);
        }
        if (wave == 0 && c + 1 < DIM / KCH_GRU)
            __builtin_amdgcn_s_wait_tensorcnt(0);
        __syncthreads();
    }
#else
    for (int k0 = 0; k0 < DIM; k0 += 32) {
        AFrag ax, ah;
        load_a_frags(xrow, hrow, k0 + kb, ax, ah);
        const __bf16* lrow = Wg + (size_t)(k0 + lane) * (FRAGS * 16);
        if (k0 + 32 < DIM) __builtin_prefetch(lrow + 32 * (FRAGS * 16), 0, 1);
        v16bf bIRf = *reinterpret_cast<const v16bf*>(lrow);
        v16bf bIZf = *reinterpret_cast<const v16bf*>(lrow + 16);
        v16bf bINf = *reinterpret_cast<const v16bf*>(lrow + 32);
        v16bf bHRf = *reinterpret_cast<const v16bf*>(lrow + 48);
        v16bf bHZf = *reinterpret_cast<const v16bf*>(lrow + 64);
        v16bf bHNf = *reinterpret_cast<const v16bf*>(lrow + 80);
        aIR = __builtin_amdgcn_wmma_f32_16x16x32_bf16(false, ax.v, false, bIRf, (short)0, aIR, false, false);
        aIZ = __builtin_amdgcn_wmma_f32_16x16x32_bf16(false, ax.v, false, bIZf, (short)0, aIZ, false, false);
        aIN = __builtin_amdgcn_wmma_f32_16x16x32_bf16(false, ax.v, false, bINf, (short)0, aIN, false, false);
        aHR = __builtin_amdgcn_wmma_f32_16x16x32_bf16(false, ah.v, false, bHRf, (short)0, aHR, false, false);
        aHZ = __builtin_amdgcn_wmma_f32_16x16x32_bf16(false, ah.v, false, bHZf, (short)0, aHZ, false, false);
        aHN = __builtin_amdgcn_wmma_f32_16x16x32_bf16(false, ah.v, false, bHNf, (short)0, aHN, false, false);
    }
#endif

    // epilogue: fused gate math. C/D layout: VGPR j -> M = m0 + j + (lane>=16)*8
    const int col = n0 + (lane & 15);
    const float bir = bIh[col], biz = bIh[DIM + col], bin = bIh[2 * DIM + col];
    const float bhr = bHh[col], bhz = bHh[DIM + col], bhn = bHh[2 * DIM + col];
    const int mbase = m0 + ((lane >> 4) << 3);
#pragma unroll
    for (int j = 0; j < 8; ++j) {
        const int m = mbase + j;
        float r = sigmoidf_(aIR[j] + bir + aHR[j] + bhr);
        float z = sigmoidf_(aIZ[j] + biz + aHZ[j] + bhz);
        float n = tanhf(aIN[j] + bin + r * (aHN[j] + bhn));
        float hp = hf[(size_t)m * DIM + col];
        float hv = (1.0f - z) * n + z * hp;
        hOut[(size_t)m * DIM + col]   = hv;
        hOutBf[(size_t)m * DIM + col] = f2bf(hv);
    }
}

// ---------------- decoder logits: [128,1024] @ [1024,8000] + bias ------------
__global__ __launch_bounds__(256)
void logits_kernel(const __bf16* __restrict__ hbf,
                   const __bf16* __restrict__ Wpack,   // [NGRP_OUT][DIM][16]
                   const float*  __restrict__ outB,
                   float* __restrict__ out) {
    const int lane = threadIdx.x & 31;
    const int wave = threadIdx.x >> 5;
    const int m0 = wave * 16;
    const int n0 = blockIdx.x * 16;

    v8f acc = {};
    const int mrow = m0 + (lane & 15);
    const int kb   = (lane >> 4) * 8;
    const __bf16* hrow = hbf + (size_t)mrow * DIM;
    const __bf16* Wg   = Wpack + (size_t)blockIdx.x * DIM * 16;

#if HAVE_TDM
    __shared__ __align__(128) __bf16 smem[2][KCH_OUT * 16];
    const unsigned ldsA = (unsigned)(uintptr_t)&smem[0][0];
    const unsigned ldsB = (unsigned)(uintptr_t)&smem[1][0];
    const unsigned CHE = KCH_OUT * 16;   // 4096 elements / 8 KB per chunk

    if (wave == 0) {
        tdm_load_1d(ldsA, Wg, CHE);
        __builtin_amdgcn_s_wait_tensorcnt(0);
    }
    __syncthreads();

    for (int c = 0; c < DIM / KCH_OUT; ++c) {
        const __bf16* curB = &smem[c & 1][0];
        if (wave == 0 && c + 1 < DIM / KCH_OUT)
            tdm_load_1d((c & 1) ? ldsA : ldsB, Wg + (size_t)(c + 1) * CHE, CHE);
#pragma unroll
        for (int kr = 0; kr < KCH_OUT; kr += 32) {
            const int k0 = c * KCH_OUT + kr;
            AFrag a;
            a.h[0] = *reinterpret_cast<const v8bf*>(hrow + k0 + kb);
            a.h[1] = *reinterpret_cast<const v8bf*>(hrow + k0 + kb + 16);
            v16bf b = *reinterpret_cast<const v16bf*>(curB + (size_t)(kr + lane) * 16);
            acc = __builtin_amdgcn_wmma_f32_16x16x32_bf16(false, a.v, false, b, (short)0, acc, false, false);
        }
        if (wave == 0 && c + 1 < DIM / KCH_OUT)
            __builtin_amdgcn_s_wait_tensorcnt(0);
        __syncthreads();
    }
#else
    for (int k0 = 0; k0 < DIM; k0 += 32) {
        AFrag a;
        a.h[0] = *reinterpret_cast<const v8bf*>(hrow + k0 + kb);
        a.h[1] = *reinterpret_cast<const v8bf*>(hrow + k0 + kb + 16);
        const __bf16* lrow = Wg + (size_t)(k0 + lane) * 16;
        if (k0 + 32 < DIM) __builtin_prefetch(lrow + 32 * 16, 0, 1);
        v16bf b = *reinterpret_cast<const v16bf*>(lrow);
        acc = __builtin_amdgcn_wmma_f32_16x16x32_bf16(false, a.v, false, b, (short)0, acc, false, false);
    }
#endif

    const int col = n0 + (lane & 15);
    const float bias = outB[col];
    const int mbase = m0 + ((lane >> 4) << 3);
#pragma unroll
    for (int j = 0; j < 8; ++j) {
        const int m = mbase + j;
        out[(size_t)m * VOUT + col] = acc[j] + bias;
    }
}

// ---------------- greedy argmax over VOUT (first-max tiebreak) ----------------
__global__ __launch_bounds__(256)
void argmax_kernel(const float* __restrict__ logits, int* __restrict__ tok) {
    const int b = blockIdx.x;
    const float* row = logits + (size_t)b * VOUT;
    float best = -3.402823466e+38f;
    int   bi   = 0x7FFFFFFF;
    for (int i = threadIdx.x; i < VOUT; i += 256) {
        float v = row[i];
        if (v > best || (v == best && i < bi)) { best = v; bi = i; }
    }
    __shared__ float sv[256];
    __shared__ int   si[256];
    sv[threadIdx.x] = best;
    si[threadIdx.x] = bi;
    __syncthreads();
    for (int s = 128; s > 0; s >>= 1) {
        if (threadIdx.x < s) {
            float v2 = sv[threadIdx.x + s];
            int   i2 = si[threadIdx.x + s];
            if (v2 > sv[threadIdx.x] || (v2 == sv[threadIdx.x] && i2 < si[threadIdx.x])) {
                sv[threadIdx.x] = v2;
                si[threadIdx.x] = i2;
            }
        }
        __syncthreads();
    }
    if (threadIdx.x == 0) tok[b] = si[0];
}

// ---------------- host orchestration ----------------
extern "C" void kernel_launch(void* const* d_in, const int* in_sizes, int n_in,
                              void* d_out, int out_size, void* d_ws, size_t ws_size,
                              hipStream_t stream) {
    (void)in_sizes; (void)n_in; (void)out_size; (void)ws_size;

    const int*   src      = (const int*)  d_in[0];
    /* tgt (d_in[1]) only provides length S_TGT = 128 */
    const float* enc_emb  = (const float*)d_in[2];
    const float* enc_W_ih = (const float*)d_in[3];
    const float* enc_W_hh = (const float*)d_in[4];
    const float* enc_b_ih = (const float*)d_in[5];
    const float* enc_b_hh = (const float*)d_in[6];
    const float* dec_emb  = (const float*)d_in[7];
    const float* dec_W_ih = (const float*)d_in[8];
    const float* dec_W_hh = (const float*)d_in[9];
    const float* dec_b_ih = (const float*)d_in[10];
    const float* dec_b_hh = (const float*)d_in[11];
    const float* out_W    = (const float*)d_in[12];
    const float* out_b    = (const float*)d_in[13];

    float* logits_out = (float*)d_out;

    // workspace carve-up (256B aligned)
    char* ws = (char*)d_ws;
    size_t off = 0;
    auto alloc = [&](size_t bytes) -> char* {
        char* p = ws + off;
        off = (off + bytes + 255) & ~(size_t)255;
        return p;
    };
    const size_t gruPackBytes = (size_t)NGRP_GRU * DIM * FRAGS * 16 * sizeof(__bf16); // 12.6 MB
    const size_t outPackBytes = (size_t)NGRP_OUT * DIM * 16 * sizeof(__bf16);         // 16.4 MB
    __bf16* encPack = (__bf16*)alloc(gruPackBytes);
    __bf16* decPack = (__bf16*)alloc(gruPackBytes);
    __bf16* outPack = (__bf16*)alloc(outPackBytes);
    float*  hF[2];
    __bf16* hB[2];
    hF[0] = (float*) alloc((size_t)BATCH * DIM * sizeof(float));
    hF[1] = (float*) alloc((size_t)BATCH * DIM * sizeof(float));
    hB[0] = (__bf16*)alloc((size_t)BATCH * DIM * sizeof(__bf16));
    hB[1] = (__bf16*)alloc((size_t)BATCH * DIM * sizeof(__bf16));
    int* tok = (int*)alloc(BATCH * sizeof(int));

    // --- one-time bf16 packed weight prep (L2-resident afterwards) ---
    {
        long long tot = (long long)NGRP_GRU * DIM * FRAGS * 16;
        int blocks = (int)((tot + 255) / 256);
        pack_gru_kernel<<<blocks, 256, 0, stream>>>(enc_W_ih, enc_W_hh, encPack);
        pack_gru_kernel<<<blocks, 256, 0, stream>>>(dec_W_ih, dec_W_hh, decPack);
        long long tot2 = (long long)NGRP_OUT * DIM * 16;
        pack_out_kernel<<<(int)((tot2 + 255) / 256), 256, 0, stream>>>(out_W, outPack);
    }

    init_kernel<<<(BATCH * DIM + 255) / 256, 256, 0, stream>>>(hF[0], hB[0], tok);
    int cur = 0;

    // --- encoder: 256 sequential GRU steps (embedding gather fused) ---
    for (int t = 0; t < S_SRC; ++t) {
        gru_step_kernel<<<NGRP_GRU, 256, 0, stream>>>(enc_emb, src + t * BATCH,
                                                      hB[cur], hF[cur], encPack,
                                                      enc_b_ih, enc_b_hh,
                                                      hF[1 - cur], hB[1 - cur]);
        cur ^= 1;
    }

    // --- decoder: 128 steps, greedy argmax feedback ---
    for (int t = 0; t < S_TGT; ++t) {
        gru_step_kernel<<<NGRP_GRU, 256, 0, stream>>>(dec_emb, tok,
                                                      hB[cur], hF[cur], decPack,
                                                      dec_b_ih, dec_b_hh,
                                                      hF[1 - cur], hB[1 - cur]);
        cur ^= 1;
        float* lg = logits_out + (size_t)t * BATCH * VOUT;
        logits_kernel<<<NGRP_OUT, 256, 0, stream>>>(hB[cur], outPack, out_b, lg);
        argmax_kernel<<<BATCH, 256, 0, stream>>>(lg, tok);
    }
}
